// SpeAttention_11914239279392
// MI455X (gfx1250) — compile-verified, matrix operand order checked by reading, not run on hardware
//
#include <hip/hip_runtime.h>
#include <hip/hip_bf16.h>
#include <math.h>

typedef __attribute__((ext_vector_type(16))) _Float16 v16h;
typedef __attribute__((ext_vector_type(8)))  float    v8f;

#define B_   32
#define C_   512
#define HW_  4096
#define DM   32
#define DI   32
#define DSN  8
#define L_   512

__device__ __forceinline__ float siluf(float x){ return x / (1.f + __expf(-x)); }
__device__ __forceinline__ float sigmf(float x){ return 1.f / (1.f + __expf(-x)); }
__device__ __forceinline__ float softplusf(float x){ return x > 20.f ? x : log1pf(__expf(x)); }
// K index for element e of a 16-bit WMMA A/B fragment (wave32 layout, half = lane>=16)
__device__ __forceinline__ int kmap(int e, int half){
  return (((e >> 1) & 3) * 2) + ((e >> 3) * 16) + half * 8 + (e & 1);
}

// ---------------- Kernel 1: adaptive avg+max pool over HW=4096 ----------------
__global__ void k_pool(const float* __restrict__ x, float* __restrict__ avg, float* __restrict__ mx){
  __shared__ float ssum[256];
  __shared__ float smax[256];
  int bc = blockIdx.x, t = threadIdx.x;
  const float4* p = (const float4*)(x + (size_t)bc * HW_);
  float s = 0.f, m = -3.4e38f;
  #pragma unroll
  for (int i = 0; i < 4; i++){
    float4 v = p[t + i * 256];
    s += v.x + v.y + v.z + v.w;
    m = fmaxf(m, fmaxf(fmaxf(v.x, v.y), fmaxf(v.z, v.w)));
  }
  ssum[t] = s; smax[t] = m; __syncthreads();
  for (int o = 128; o > 0; o >>= 1){
    if (t < o){ ssum[t] += ssum[t + o]; smax[t] = fmaxf(smax[t], smax[t + o]); }
    __syncthreads();
  }
  if (t == 0){ avg[bc] = ssum[0] * (1.f / HW_); mx[bc] = smax[0]; }
}

// ---------------- Kernel 2: lin1 + layernorm -> f16 tokens ----------------
__global__ void k_embed(const float* __restrict__ avgb, const float* __restrict__ mxb,
                        const float* __restrict__ w1, const float* __restrict__ b1,
                        const float* __restrict__ g, const float* __restrict__ bt,
                        _Float16* __restrict__ h16){
  int idx = blockIdx.x * 256 + threadIdx.x;        // token (b*512 + l)
  float a = avgb[idx], m = mxb[idx];
  float h[DM];
  float mean = 0.f;
  #pragma unroll
  for (int c = 0; c < DM; c++){
    h[c] = a * w1[2*c] + m * w1[2*c + 1] + b1[c];
    mean += h[c];
  }
  mean *= (1.f / DM);
  float var = 0.f;
  #pragma unroll
  for (int c = 0; c < DM; c++){ float d = h[c] - mean; var += d * d; }
  var *= (1.f / DM);
  float r = rsqrtf(var + 1e-5f);
  #pragma unroll
  for (int c = 0; c < DM; c++)
    h16[(size_t)idx * DM + c] = (_Float16)((h[c] - mean) * r * g[c] + bt[c]);
}

// ---------------- Kernel 3: convert projection weights to f16 (transposed K x N) ----------------
__global__ void k_prep(const float* __restrict__ inw, const float* __restrict__ outw,
                       _Float16* __restrict__ inT, _Float16* __restrict__ outT){
  int t = threadIdx.x;
  for (int i = t; i < 64 * 32; i += 256){ int e = i >> 5, c = i & 31; inT[c * 64 + e] = (_Float16)inw[i]; }
  for (int i = t; i < 32 * 32; i += 256){ int o = i >> 5, d = i & 31; outT[d * 32 + o] = (_Float16)outw[i]; }
}

// ---------------- Kernel 4: xz = h @ in_projT  (WMMA f16 K=32), out [b][64][512] ----------------
__global__ void k_inproj(const _Float16* __restrict__ h16, const _Float16* __restrict__ wT,
                         float* __restrict__ xz){
  int tid  = threadIdx.x;
  int lane = tid & 31;
  int gw   = blockIdx.x * 8 + (tid >> 5);   // global wave id
  int b    = gw >> 7;                       // 128 tiles per batch (32 M-tiles x 4 N-tiles)
  int tile = gw & 127;
  int mt = tile >> 2, nt = tile & 3;
  int half = lane >> 4, mr = lane & 15;
  v16h a, bf;
  #pragma unroll
  for (int e = 0; e < 16; e++){
    int k = kmap(e, half);
    a[e]  = h16[(size_t)((b * L_ + mt * 16 + mr) * DM) + k];
    bf[e] = wT[k * 64 + nt * 16 + mr];
  }
  v8f c = {};
  c = __builtin_amdgcn_wmma_f32_16x16x32_f16(false, a, false, bf, (short)0, c, false, false);
  int ncol = nt * 16 + mr;
  #pragma unroll
  for (int r = 0; r < 8; r++){
    int tok = mt * 16 + r + 8 * half;
    xz[(size_t)((b * 64 + ncol) * L_) + tok] = c[r];
  }
}

// ---------------- Kernel 5: conv+silu, xproj, dtproj+softplus, silu(z); scan-ordered ----------------
__global__ void k_pre(const float* __restrict__ xz,
                      const float* cwf, const float* cbf, const float* xpf, const float* dwf, const float* dbf,
                      const float* cwr, const float* cbr, const float* xpr, const float* dwr, const float* dbr,
                      float* __restrict__ dtb, float* __restrict__ xhb, float* __restrict__ zgb,
                      float* __restrict__ Bb, float* __restrict__ Cb){
  int blk = blockIdx.x;
  int b   = blk >> 2;
  int dir = (blk >> 1) & 1;
  int ls  = ((blk & 1) << 8) + threadIdx.x;        // scan-order position
  const float* cw = dir ? cwr : cwf;
  const float* cb = dir ? cbr : cbf;
  const float* xp = dir ? xpr : xpf;
  const float* dw = dir ? dwr : dwf;
  const float* db = dir ? dbr : dbf;
  const float* xzb = xz + (size_t)b * 64 * L_;
  int base = b * 2 + dir;

  float xc[DI];
  #pragma unroll
  for (int d = 0; d < DI; d++){
    float acc = cb[d];
    #pragma unroll
    for (int k = 0; k < 4; k++){
      int j = ls - 3 + k;                          // causal pad with zeros
      if (j >= 0){
        int jj = dir ? (L_ - 1 - j) : j;           // reverse dir reads flipped sequence
        acc += cw[d * 4 + k] * xzb[(size_t)d * L_ + jj];
      }
    }
    xc[d] = siluf(acc);
  }
  float dt0 = 0.f, dt1 = 0.f;
  #pragma unroll
  for (int d = 0; d < DI; d++){ dt0 += xp[d] * xc[d]; dt1 += xp[DI + d] * xc[d]; }
  #pragma unroll
  for (int j = 0; j < DSN; j++){
    float s = 0.f;
    #pragma unroll
    for (int d = 0; d < DI; d++) s += xp[(2 + j) * DI + d] * xc[d];
    Bb[(size_t)(base * DSN + j) * L_ + ls] = s;
  }
  #pragma unroll
  for (int j = 0; j < DSN; j++){
    float s = 0.f;
    #pragma unroll
    for (int d = 0; d < DI; d++) s += xp[(10 + j) * DI + d] * xc[d];
    Cb[(size_t)(base * DSN + j) * L_ + ls] = s;
  }
  int lsrc = dir ? (L_ - 1 - ls) : ls;
  #pragma unroll
  for (int d = 0; d < DI; d++){
    float dtv = softplusf(dw[d * 2] * dt0 + dw[d * 2 + 1] * dt1 + db[d]);
    dtb[(size_t)(base * DI + d) * L_ + ls] = dtv;
    xhb[(size_t)(base * DI + d) * L_ + ls] = xc[d];
    float zv = xzb[(size_t)(DI + d) * L_ + lsrc];
    zgb[(size_t)(base * DI + d) * L_ + ls] = siluf(zv);
  }
}

// ---------------- Kernel 6: selective scan; thread = (d,n) state element ----------------
__global__ void k_scan(const float* __restrict__ dtb, const float* __restrict__ xhb,
                       const float* __restrict__ zgb, const float* __restrict__ Bb,
                       const float* __restrict__ Cb,
                       const float* Alf, const float* Df, const float* Alr, const float* Dr,
                       float* __restrict__ yb){
  int b   = blockIdx.x >> 1;
  int dir = blockIdx.x & 1;
  int t = threadIdx.x;
  int d = t >> 3, n = t & 7;
  const float* Al = dir ? Alr : Alf;
  const float* Dd = dir ? Dr  : Df;
  int base = b * 2 + dir;
  const float* dtrow = dtb + (size_t)(base * DI + d) * L_;
  const float* xhrow = xhb + (size_t)(base * DI + d) * L_;
  const float* zrow  = zgb + (size_t)(base * DI + d) * L_;
  const float* Brow  = Bb  + (size_t)(base * DSN + n) * L_;
  const float* Crow  = Cb  + (size_t)(base * DSN + n) * L_;
  float* yrow = yb + (size_t)(base * DI + d) * L_;
  float A  = -__expf(Al[d * DSN + n]);
  float Dv = Dd[d];
  float h = 0.f;
  for (int ls = 0; ls < L_; ls++){
    float dt = dtrow[ls], u = xhrow[ls];
    h = __expf(dt * A) * h + dt * Brow[ls] * u;
    float p = h * Crow[ls];
    p += __shfl_xor(p, 1);   // reduce over n within 8 contiguous lanes (wave32)
    p += __shfl_xor(p, 2);
    p += __shfl_xor(p, 4);
    if (n == 0) yrow[ls] = (p + Dv * u) * zrow[ls];
  }
}

// ---------------- Kernel 7: combine dirs, out_proj (WMMA), LN, sigmoid gate ----------------
__global__ void k_out(const float* __restrict__ yb, const _Float16* __restrict__ wT,
                      const float* __restrict__ mg, const float* __restrict__ mb,
                      const float* __restrict__ l2w, const float* __restrict__ l2b,
                      float* __restrict__ out){
  __shared__ float so[64][33];
  int tid  = threadIdx.x;        // 128 threads = 4 waves
  int lane = tid & 31;
  int w    = tid >> 5;
  int b = blockIdx.x >> 3;
  int g = blockIdx.x & 7;
  int l0 = g * 64;
  int half = lane >> 4, mr = lane & 15;
  const float* yf = yb + (size_t)(b * 2 + 0) * DI * L_;
  const float* yr = yb + (size_t)(b * 2 + 1) * DI * L_;
  int l = l0 + w * 16 + mr;
  v16h a;
  #pragma unroll
  for (int e = 0; e < 16; e++){
    int k = kmap(e, half);       // k = d channel
    a[e] = (_Float16)(0.5f * (yf[(size_t)k * L_ + l] + yr[(size_t)k * L_ + (L_ - 1 - l)]));
  }
  #pragma unroll
  for (int nt = 0; nt < 2; nt++){
    v16h bf;
    #pragma unroll
    for (int e = 0; e < 16; e++){
      int k = kmap(e, half);
      bf[e] = wT[k * DM + nt * 16 + mr];
    }
    v8f c = {};
    c = __builtin_amdgcn_wmma_f32_16x16x32_f16(false, a, false, bf, (short)0, c, false, false);
    #pragma unroll
    for (int r = 0; r < 8; r++)
      so[w * 16 + r + 8 * half][nt * 16 + mr] = c[r];
  }
  __syncthreads();
  if (tid < 64){
    int lo = l0 + tid;
    float mean = 0.f;
    #pragma unroll
    for (int o = 0; o < DM; o++) mean += so[tid][o];
    mean *= (1.f / DM);
    float var = 0.f;
    #pragma unroll
    for (int o = 0; o < DM; o++){ float dd = so[tid][o] - mean; var += dd * dd; }
    var *= (1.f / DM);
    float rr = rsqrtf(var + 1e-5f);
    float acc = l2b[0];
    #pragma unroll
    for (int o = 0; o < DM; o++){
      float nv = (so[tid][o] - mean) * rr * mg[o] + mb[o];
      acc += nv * l2w[o];
    }
    out[b * C_ + lo] = sigmf(acc) * 0.25f + 0.875f;
  }
}

extern "C" void kernel_launch(void* const* d_in, const int* in_sizes, int n_in,
                              void* d_out, int out_size, void* d_ws, size_t ws_size,
                              hipStream_t stream) {
  (void)in_sizes; (void)n_in; (void)out_size; (void)ws_size;
  const float* x         = (const float*)d_in[0];
  const float* lin1_w    = (const float*)d_in[1];
  const float* lin1_b    = (const float*)d_in[2];
  const float* ln1_g     = (const float*)d_in[3];
  const float* ln1_b     = (const float*)d_in[4];
  const float* in_proj_w = (const float*)d_in[5];
  const float* conv_w_f  = (const float*)d_in[6];
  const float* conv_b_f  = (const float*)d_in[7];
  const float* xproj_w_f = (const float*)d_in[8];
  const float* dtproj_w_f= (const float*)d_in[9];
  const float* dtproj_b_f= (const float*)d_in[10];
  const float* A_log_f   = (const float*)d_in[11];
  const float* D_f       = (const float*)d_in[12];
  const float* conv_w_r  = (const float*)d_in[13];
  const float* conv_b_r  = (const float*)d_in[14];
  const float* xproj_w_r = (const float*)d_in[15];
  const float* dtproj_w_r= (const float*)d_in[16];
  const float* dtproj_b_r= (const float*)d_in[17];
  const float* A_log_r   = (const float*)d_in[18];
  const float* D_r       = (const float*)d_in[19];
  const float* out_proj_w= (const float*)d_in[20];
  const float* mnorm_g   = (const float*)d_in[21];
  const float* mnorm_b   = (const float*)d_in[22];
  const float* lin2_w    = (const float*)d_in[23];
  const float* lin2_b    = (const float*)d_in[24];
  float* outp = (float*)d_out;

  char* wsb = (char*)d_ws;
  size_t off = 0;
  auto take = [&](size_t bytes) -> void* {
    void* p = (void*)(wsb + off);
    off += (bytes + 255) & ~(size_t)255;
    return p;
  };
  float*     avg  = (float*)    take((size_t)B_ * C_ * 4);
  float*     mxp  = (float*)    take((size_t)B_ * C_ * 4);
  _Float16*  h16  = (_Float16*) take((size_t)B_ * C_ * DM * 2);
  _Float16*  inT  = (_Float16*) take((size_t)64 * 32 * 2);
  _Float16*  outT = (_Float16*) take((size_t)32 * 32 * 2);
  float*     xz   = (float*)    take((size_t)B_ * 64 * L_ * 4);
  float*     dtb  = (float*)    take((size_t)B_ * 2 * DI * L_ * 4);
  float*     xhb  = (float*)    take((size_t)B_ * 2 * DI * L_ * 4);
  float*     zgb  = (float*)    take((size_t)B_ * 2 * DI * L_ * 4);
  float*     Bb   = (float*)    take((size_t)B_ * 2 * DSN * L_ * 4);
  float*     Cb   = (float*)    take((size_t)B_ * 2 * DSN * L_ * 4);
  float*     yb   = (float*)    take((size_t)B_ * 2 * DI * L_ * 4);

  k_pool  <<<B_ * C_, 256, 0, stream>>>(x, avg, mxp);
  k_embed <<<(B_ * C_) / 256, 256, 0, stream>>>(avg, mxp, lin1_w, lin1_b, ln1_g, ln1_b, h16);
  k_prep  <<<1, 256, 0, stream>>>(in_proj_w, out_proj_w, inT, outT);
  k_inproj<<<512, 256, 0, stream>>>(h16, inT, xz);
  k_pre   <<<128, 256, 0, stream>>>(xz,
                                    conv_w_f, conv_b_f, xproj_w_f, dtproj_w_f, dtproj_b_f,
                                    conv_w_r, conv_b_r, xproj_w_r, dtproj_w_r, dtproj_b_r,
                                    dtb, xhb, zgb, Bb, Cb);
  k_scan  <<<B_ * 2, 256, 0, stream>>>(dtb, xhb, zgb, Bb, Cb, A_log_f, D_f, A_log_r, D_r, yb);
  k_out   <<<B_ * 8, 128, 0, stream>>>(yb, outT, mnorm_g, mnorm_b, lin2_w, lin2_b, outp);
}
